// HardNetLoss_2576980377654
// MI455X (gfx1250) — matrix operand order repaired and checked by reading, not compiled, not run on hardware
//
#include <hip/hip_runtime.h>
#include <hip/hip_bf16.h>
#include <math.h>

typedef __attribute__((ext_vector_type(16))) _Float16 v16h;
typedef __attribute__((ext_vector_type(8)))  _Float16 v8h;
typedef __attribute__((ext_vector_type(4)))  _Float16 v4h;
typedef __attribute__((ext_vector_type(8)))  float    v8f;

#define CNT      8192
#define KDIM     256
#define JT_TOTAL 512              // CNT / 16
#define SPLIT    16
#define JT_PER   (JT_TOTAL / SPLIT)   // 32 column tiles per block
#define LDS_ROW_H 272             // 256 halves + 16 pad (544 B, 32B-aligned rows)

// Monotone float<->uint key: ascending float order == ascending uint order.
__device__ __forceinline__ unsigned fkey(float f) {
  unsigned u = __float_as_uint(f);
  return (u & 0x80000000u) ? ~u : (u | 0x80000000u);
}
__device__ __forceinline__ float funkey(unsigned k) {
  unsigned u = (k & 0x80000000u) ? (k & 0x7FFFFFFFu) : ~k;
  return __uint_as_float(u);
}

// ---------------- fp32 -> f16 conversion ----------------
__global__ __launch_bounds__(256) void hardnet_cvt_kernel(
    const float* __restrict__ x, _Float16* __restrict__ o, int n4) {
  int t = blockIdx.x * blockDim.x + threadIdx.x;
  if (t < n4) {
    float4 f = ((const float4*)x)[t];
    v4h h;
    h.x = (_Float16)f.x; h.y = (_Float16)f.y;
    h.z = (_Float16)f.z; h.w = (_Float16)f.w;
    ((v4h*)o)[t] = h;
  }
}

// ---------------- init row-max key arrays ----------------
__global__ __launch_bounds__(256) void hardnet_init_kernel(
    unsigned* __restrict__ u, int n) {
  int t = blockIdx.x * blockDim.x + threadIdx.x;
  if (t < n) u[t] = 0u;  // smallest monotone key == -inf
}

// ---------------- fused WMMA GEMM + row-max of s (+diag s) ----------------
// d = sqrt(2(1-s+eps)) is monotone-decreasing in s, so min_row(d) == d(max_row(s)).
// Each wave register-blocks TWO 16-row A strips: every B fragment loaded from
// LDS feeds two independent WMMAs (16 ds_load : 16 wmma per tile).
__global__ __launch_bounds__(256) void hardnet_gemm_max_kernel(
    const _Float16* __restrict__ Arows,
    const _Float16* __restrict__ Prows,
    unsigned* __restrict__ maxKey,
    float* __restrict__ posS,
    int writePos) {
  __shared__ _Float16 Btile[2][16 * LDS_ROW_H] __attribute__((aligned(32)));

  const int tid  = threadIdx.x;
  const int lane = tid & 31;
  const int wave = tid >> 5;
  const int sel  = lane >> 4;   // half-wave select
  const int lr   = lane & 15;   // row-in-frag (A) / col-in-frag (B, C/D)
  const int rowbase = (blockIdx.x * 8 + wave) * 32;   // 32 rows per wave
  // Wave-uniform copy for *scalar* diag-tile branches (s_cmp/s_cbranch).
  const int rowbase_u = __builtin_amdgcn_readfirstlane(rowbase);

  // Hoist two A strips (2 x 16 rows x 256 K) as 16 WMMA fragments (128 VGPRs).
  // A fragments use contiguous K order here (B matches): lanes 0-15 hold K {k0..k0+15},
  // lanes 16-31 hold K {k0+16..k0+31}.
  const _Float16* arow0 = Arows + (size_t)(rowbase + lr) * KDIM;
  const _Float16* arow1 = arow0 + (size_t)16 * KDIM;
  v16h Afrag0[8], Afrag1[8];
#pragma unroll
  for (int kk = 0; kk < 8; ++kk) {
    v8h lo0 = *(const v8h*)(arow0 + kk * 32 + sel * 16);
    v8h hi0 = *(const v8h*)(arow0 + kk * 32 + sel * 16 + 8);
    v8h lo1 = *(const v8h*)(arow1 + kk * 32 + sel * 16);
    v8h hi1 = *(const v8h*)(arow1 + kk * 32 + sel * 16 + 8);
    v16h a0, a1;
#pragma unroll
    for (int i = 0; i < 8; ++i) {
      a0[i] = lo0[i]; a0[i + 8] = hi0[i];
      a1[i] = lo1[i]; a1[i + 8] = hi1[i];
    }
    Afrag0[kk] = a0;
    Afrag1[kk] = a1;
  }

  float rowmax0[8], rowmax1[8];
#pragma unroll
  for (int v = 0; v < 8; ++v) { rowmax0[v] = -3.0e38f; rowmax1[v] = -3.0e38f; }

  const int jt0 = blockIdx.y * JT_PER;
  const int jtEnd = jt0 + JT_PER;

  // Cooperative fill helpers: 16 P-rows are a contiguous 8 KB chunk.
  const int c0i = tid, c1i = tid + 256;
  const int do0 = (c0i >> 5) * (LDS_ROW_H * 2) + (c0i & 31) * 16;
  const int do1 = (c1i >> 5) * (LDS_ROW_H * 2) + (c1i & 31) * 16;

  // Preamble: fill buffer 0 with tile jt0.
  {
    const float4* src = (const float4*)(Prows + (size_t)jt0 * 16 * KDIM);
    char* dst = (char*)Btile[0];
    *(float4*)(dst + do0) = src[c0i];
    *(float4*)(dst + do1) = src[c1i];
  }

  int buf = 0;
  for (int jt = jt0; jt < jtEnd; ++jt) {
    const int colbase = jt * 16;
    __syncthreads();   // Btile[buf] for jt is ready; Btile[buf^1] free

    // Issue next tile's global loads BEFORE the WMMA chain (latency hidden).
    const bool havenext = (jt + 1 < jtEnd);
    float4 r0, r1;
    if (havenext) {
      const float4* src = (const float4*)(Prows + (size_t)(colbase + 16) * KDIM);
      r0 = src[c0i];
      r1 = src[c1i];
      if (jt + 2 < jtEnd)
        __builtin_prefetch(Prows + (size_t)(colbase + 32) * KDIM, 0, 1);
    }

    v8f c0 = {}, c1 = {};
#pragma unroll
    for (int kk = 0; kk < 8; ++kk) {
      // B 32x16 f16 fragment: lane = column N, contiguous K halves per half-wave.
      const v16h* bp = (const v16h*)((const char*)Btile[buf] +
                                     lr * (LDS_ROW_H * 2) + kk * 64 + sel * 32);
      v16h b = *bp;
      // Two independent WMMAs share one B fragment (register-level B reuse).
      c0 = __builtin_amdgcn_wmma_f32_16x16x32_f16(
          false, Afrag0[kk], false, b, (short)0, c0, false, false);
      c1 = __builtin_amdgcn_wmma_f32_16x16x32_f16(
          false, Afrag1[kk], false, b, (short)0, c1, false, false);
    }

    // Strip 0 epilogue (rows rowbase..rowbase+15)
    if (colbase == rowbase_u) {
#pragma unroll
      for (int v = 0; v < 8; ++v) {
        float s = c0[v];
        int m = v + sel * 8;              // C/D layout: VGPR v -> M=v(+8)
        bool diag = (m == lr);
        if (diag && writePos) posS[rowbase + m] = s;
        rowmax0[v] = fmaxf(rowmax0[v], diag ? -3.0e38f : s);
      }
    } else {
#pragma unroll
      for (int v = 0; v < 8; ++v) rowmax0[v] = fmaxf(rowmax0[v], c0[v]);
    }
    // Strip 1 epilogue (rows rowbase+16..rowbase+31)
    if (colbase == rowbase_u + 16) {
#pragma unroll
      for (int v = 0; v < 8; ++v) {
        float s = c1[v];
        int m = v + sel * 8;
        bool diag = (m == lr);
        if (diag && writePos) posS[rowbase + 16 + m] = s;
        rowmax1[v] = fmaxf(rowmax1[v], diag ? -3.0e38f : s);
      }
    } else {
#pragma unroll
      for (int v = 0; v < 8; ++v) rowmax1[v] = fmaxf(rowmax1[v], c1[v]);
    }

    // Stage next tile into the other buffer after compute.
    if (havenext) {
      char* dst = (char*)Btile[buf ^ 1];
      *(float4*)(dst + do0) = r0;
      *(float4*)(dst + do1) = r1;
    }
    buf ^= 1;
  }

  // Row-max across the 16 lanes of each half-wave, then monotone-key atomicMax.
#pragma unroll
  for (int v = 0; v < 8; ++v) {
    float m0 = rowmax0[v];
    m0 = fmaxf(m0, __shfl_xor(m0, 1, 32));
    m0 = fmaxf(m0, __shfl_xor(m0, 2, 32));
    m0 = fmaxf(m0, __shfl_xor(m0, 4, 32));
    m0 = fmaxf(m0, __shfl_xor(m0, 8, 32));
    float m1 = rowmax1[v];
    m1 = fmaxf(m1, __shfl_xor(m1, 1, 32));
    m1 = fmaxf(m1, __shfl_xor(m1, 2, 32));
    m1 = fmaxf(m1, __shfl_xor(m1, 4, 32));
    m1 = fmaxf(m1, __shfl_xor(m1, 8, 32));
    if (lr == 0) {
      atomicMax(&maxKey[rowbase + sel * 8 + v], fkey(m0));
      atomicMax(&maxKey[rowbase + 16 + sel * 8 + v], fkey(m1));
    }
  }
}

// ---------------- final scalar reduction (sqrt applied here) ----------------
__global__ __launch_bounds__(256) void hardnet_reduce_kernel(
    const unsigned* __restrict__ maxP, const unsigned* __restrict__ maxA,
    const float* __restrict__ posS, float* __restrict__ out) {
  __shared__ float sm[256];
  int tid = threadIdx.x;
  float acc = 0.0f;
  for (int i = tid; i < CNT; i += 256) {
    float smax = fmaxf(funkey(maxA[i]), funkey(maxP[i]));
    float neg = sqrtf(fmaxf(2.0f * (1.0f - smax + 1e-6f), 0.0f));
    float pos = sqrtf(fmaxf(2.0f * (1.0f - posS[i] + 1e-6f), 0.0f));
    acc += fmaxf(0.0f, 1.0f - neg + pos);
  }
  sm[tid] = acc;
  __syncthreads();
  for (int off = 128; off > 0; off >>= 1) {
    if (tid < off) sm[tid] += sm[tid + off];
    __syncthreads();
  }
  if (tid == 0) out[0] = sm[0] / (float)CNT;
}

extern "C" void kernel_launch(void* const* d_in, const int* in_sizes, int n_in,
                              void* d_out, int out_size, void* d_ws, size_t ws_size,
                              hipStream_t stream) {
  (void)in_sizes; (void)n_in; (void)out_size; (void)ws_size;
  const float* x = (const float*)d_in[0];

  // workspace layout: [Ah 4MB][Ph 4MB][maxP 32KB][maxA 32KB][posS 32KB]
  _Float16* Ah = (_Float16*)d_ws;
  _Float16* Ph = Ah + (size_t)CNT * KDIM;
  unsigned* maxP = (unsigned*)(Ph + (size_t)CNT * KDIM);
  unsigned* maxA = maxP + CNT;
  float*    posS = (float*)(maxA + CNT);

  const int n4 = (2 * CNT * KDIM) / 4;
  hardnet_cvt_kernel<<<n4 / 256, 256, 0, stream>>>(x, Ah, n4);
  hardnet_init_kernel<<<(2 * CNT + 255) / 256, 256, 0, stream>>>(maxP, 2 * CNT);

  // 32 row-groups (8 waves x 32 rows each) x 16 j-splits = 512 blocks
  dim3 grid(CNT / 256, SPLIT);
  // pass 1: rows of a.p^T -> row maxes of s (== min_p after transform) + diag s
  hardnet_gemm_max_kernel<<<grid, 256, 0, stream>>>(Ah, Ph, maxP, posS, 1);
  // pass 2: rows of p.a^T -> row maxes of s^T == column maxes (== min_a)
  hardnet_gemm_max_kernel<<<grid, 256, 0, stream>>>(Ph, Ah, maxA, posS, 0);

  hardnet_reduce_kernel<<<1, 256, 0, stream>>>(maxP, maxA, posS, (float*)d_out);
}
